// OuroLayer_46462956208931
// MI455X (gfx1250) — matrix operand order repaired and checked by reading, not compiled, not run on hardware
//
#include <hip/hip_runtime.h>
#include <math.h>

typedef __attribute__((ext_vector_type(16))) _Float16 v16h;
typedef __attribute__((ext_vector_type(8)))  float    v8f;

#define DEVINL __device__ __forceinline__

// ---------------------------------------------------------------------------
// CDNA5 async global->LDS copy (ASYNCcnt path, CDNA5 ISA)
// ---------------------------------------------------------------------------
DEVINL void async_g2l_b128(const float* gptr, const float* lptr) {
  unsigned loff = (unsigned)(unsigned long long)lptr;  // LDS byte offset (addr[31:0])
  asm volatile("global_load_async_to_lds_b128 %0, %1, off"
               :: "v"(loff), "v"(gptr) : "memory");
}
DEVINL void wait_async0() { asm volatile("s_wait_asynccnt 0x0" ::: "memory"); }

// ---------------------------------------------------------------------------
// WMMA operand layouts (CDNA5 ISA 7.12.2, wave32)
// A 16x32 f16:  lane L: row = L%16, half = L/16; elem j -> k = 16*half + j (same order as B)
// B 32x16 f16:  lane L: col n = L%16, half = L/16; elem j -> k = 16*half + j (order this code uses for A and B)
// C/D 16x16 f32: lane L: col = L%16; VGPR i -> row = 8*(L/16) + i
// ---------------------------------------------------------------------------
DEVINL v16h loadA16(const float* __restrict__ base, int ld, int row, int k0, int half) {
  const float* p = base + (size_t)row * ld + k0 + 16 * half;
  v16h a;
#pragma unroll
  for (int j = 0; j < 8; ++j) a[j] = (_Float16)p[j];
#pragma unroll
  for (int j = 0; j < 8; ++j) a[j + 8] = (_Float16)p[8 + j];
  return a;
}

DEVINL v16h loadB_nk(const float* __restrict__ W, int ld, int n, int k0, int half) {
  const float* p = W + (size_t)n * ld + k0 + 16 * half;
  v16h b;
#pragma unroll
  for (int j = 0; j < 16; ++j) b[j] = (_Float16)p[j];
  return b;
}

// ---------------------------------------------------------------------------
// Blocked WMMA GEMM: out[b] = epi(alpha * (A[b] @ op(W[b])^T) + bias)
//   Block tile 64(M) x 128(N); 4 waves, wave w owns cols [w*32, w*32+31].
//   A block (64x32) staged in LDS as f16 (converted once at store);
//   W block (f32, 16KB) staged via global_load_async_to_lds_b128 (ASYNCcnt),
//   which also coalesces the strided [K,N] weight layouts (WKN==1).
//   8 WMMA per wave per k-step; B registers reused across 4 M subtiles.
//   Template params make each specialization a straight-line hot loop.
// grid = (N/128, M/64, nbatch), block = 128
// ---------------------------------------------------------------------------
template <int WKN, int ACT, bool HB, bool HR, bool CAUSAL>
__global__ __launch_bounds__(128) void gemm_wmma(
    const float* __restrict__ A, long long sAb, int lda,
    const float* __restrict__ W, long long sWb, int ldw,
    const float* __restrict__ bias,
    const float* __restrict__ res, long long sRb, int ldr,
    float* __restrict__ out, long long sOb, int ldo,
    int K, float alpha)
{
  __shared__ __align__(16) float    LW[32 * 128];   // 16KB W block tile (f32)
  __shared__ __align__(16) _Float16 LAs[64 * 32];   // 4KB  A block tile (f16)

  const int tid = threadIdx.x;
  const int lane = tid & 31;
  const int wv = tid >> 5;            // wave id 0..3 -> 32-col group
  const int r = lane & 15;
  const int half = lane >> 4;
  const int row0 = blockIdx.y << 6;   // 64-row block
  const int col0 = blockIdx.x << 7;   // 128-col block

  A += (size_t)blockIdx.z * sAb;
  W += (size_t)blockIdx.z * sWb;
  out += (size_t)blockIdx.z * sOb;
  if constexpr (HR) res += (size_t)blockIdx.z * sRb;

  const int c0 = wv * 32 + r;         // lane's col (LDS-local) for subtile 0
  const int c1 = c0 + 16;             // subtile 1

  v8f acc[8] = {};                    // [ms*2 + sub]

  // A staging role for this thread: 16 contiguous f32 of one half-row
  const int arow = tid >> 1;
  const int akh = (tid & 1) * 16;

  for (int k0 = 0; k0 < K; k0 += 32) {
    __syncthreads();  // WAR: previous iteration's LDS reads complete

    // --- stage W block via async copy (no VGPR round-trip) ---
#pragma unroll
    for (int c = 0; c < 8; ++c) {
      const int f = tid * 8 + c;      // 1024 float4 chunks (16KB)
      if constexpr (WKN == 0) {       // W[N,K]: LW[128][32]
        const int wrow = f >> 3, kk = (f & 7) << 2;
        async_g2l_b128(W + (size_t)(col0 + wrow) * ldw + k0 + kk,
                       &LW[wrow * 32 + kk]);
      } else {                        // W[K,N]: LW[32][128]
        const int kr = f >> 5, cc = (f & 31) << 2;
        async_g2l_b128(W + (size_t)(k0 + kr) * ldw + col0 + cc,
                       &LW[kr * 128 + cc]);
      }
    }

    // --- stage A block as f16 (convert once), overlapping the async W ---
    {
      const float* gp = A + (size_t)(row0 + arow) * lda + k0 + akh;
      if (k0 + 32 < K) __builtin_prefetch(gp + 32, 0, 0);  // global_prefetch_b8
      v16h av;
#pragma unroll
      for (int j = 0; j < 16; ++j) av[j] = (_Float16)gp[j];
      *(v16h*)&LAs[arow * 32 + akh] = av;
    }

    wait_async0();
    __syncthreads();

    // --- B operands for this wave's two 16-col subtiles (reused over 4 ms) ---
    v16h b0, b1;
    if constexpr (WKN == 0) {
      const float* p0 = &LW[c0 * 32 + 16 * half];
      const float* p1 = &LW[c1 * 32 + 16 * half];
#pragma unroll
      for (int j = 0; j < 16; ++j) { b0[j] = (_Float16)p0[j]; b1[j] = (_Float16)p1[j]; }
    } else {
#pragma unroll
      for (int j = 0; j < 16; ++j) {
        b0[j] = (_Float16)LW[(16 * half + j) * 128 + c0];
        b1[j] = (_Float16)LW[(16 * half + j) * 128 + c1];
      }
    }

#pragma unroll
    for (int ms = 0; ms < 4; ++ms) {
      // A operand: two 16B ds loads of ready f16, no conversion
      const _Float16* lp = &LAs[(ms * 16 + r) * 32 + 16 * half];
      v16h a;
#pragma unroll
      for (int j = 0; j < 8; ++j) a[j] = lp[j];
#pragma unroll
      for (int j = 0; j < 8; ++j) a[8 + j] = lp[8 + j];
      acc[ms * 2 + 0] = __builtin_amdgcn_wmma_f32_16x16x32_f16(false, a, false, b0, (short)0, acc[ms * 2 + 0], false, false);
      acc[ms * 2 + 1] = __builtin_amdgcn_wmma_f32_16x16x32_f16(false, a, false, b1, (short)0, acc[ms * 2 + 1], false, false);
    }
  }

#pragma unroll
  for (int sub = 0; sub < 2; ++sub) {
    const int col = col0 + wv * 32 + sub * 16 + r;
    float bv = 0.0f;
    if constexpr (HB) bv = bias[col];
#pragma unroll
    for (int ms = 0; ms < 4; ++ms) {
      v8f ac = acc[ms * 2 + sub];
#pragma unroll
      for (int i = 0; i < 8; ++i) {
        const int row = row0 + ms * 16 + 8 * half + i;
        float v = ac[i] * alpha + bv;
        if constexpr (ACT == 1)      v = v * (1.0f / (1.0f + __expf(-v)));   // silu
        else if constexpr (ACT == 2) v = 1.0f / (1.0f + __expf(-v));         // sigmoid
        if constexpr (HR) v += res[(size_t)row * ldr + col];
        if constexpr (CAUSAL) { if (col > row) v = 0.0f; }
        out[(size_t)row * ldo + col] = v;
      }
    }
  }
}

// ---------------------------------------------------------------------------
// LayerNorm, one block per row (C elements)
// ---------------------------------------------------------------------------
__global__ __launch_bounds__(256) void ln_kernel(
    const float* __restrict__ x, const float* __restrict__ g,
    const float* __restrict__ beta, float* __restrict__ out, int C)
{
  const int row = blockIdx.x;
  const float* xr = x + (size_t)row * C;
  float* orow = out + (size_t)row * C;
  float s = 0.f, s2 = 0.f;
  for (int i = threadIdx.x; i < C; i += 256) { float v = xr[i]; s += v; s2 += v * v; }
#pragma unroll
  for (int m = 16; m >= 1; m >>= 1) { s += __shfl_xor(s, m, 32); s2 += __shfl_xor(s2, m, 32); }
  __shared__ float sm[8], sm2[8];
  if ((threadIdx.x & 31) == 0) { sm[threadIdx.x >> 5] = s; sm2[threadIdx.x >> 5] = s2; }
  __syncthreads();
  s = 0.f; s2 = 0.f;
#pragma unroll
  for (int i = 0; i < 8; ++i) { s += sm[i]; s2 += sm2[i]; }
  const float mean = s / C;
  const float inv = rsqrtf(s2 / C - mean * mean + 1e-5f);
  for (int i = threadIdx.x; i < C; i += 256)
    orow[i] = (xr[i] - mean) * inv * g[i] + beta[i];
}

// Row L2-normalize in place: v /= max(||v||, 1e-5)
__global__ __launch_bounds__(256) void rownorm_kernel(float* __restrict__ v, int C)
{
  const int row = blockIdx.x;
  float* vr = v + (size_t)row * C;
  float s2 = 0.f;
  for (int i = threadIdx.x; i < C; i += 256) { float x = vr[i]; s2 += x * x; }
#pragma unroll
  for (int m = 16; m >= 1; m >>= 1) s2 += __shfl_xor(s2, m, 32);
  __shared__ float sm[8];
  if ((threadIdx.x & 31) == 0) sm[threadIdx.x >> 5] = s2;
  __syncthreads();
  s2 = 0.f;
#pragma unroll
  for (int i = 0; i < 8; ++i) s2 += sm[i];
  const float sc = 1.0f / fmaxf(sqrtf(s2), 1e-5f);
  for (int i = threadIdx.x; i < C; i += 256) vr[i] *= sc;
}

// ---------------------------------------------------------------------------
// RoPE + head layout: qkv[B,T,3C] -> Qr/Kr/Vr [B*H, T, HD] (rope on Q,K)
// ---------------------------------------------------------------------------
__global__ __launch_bounds__(256) void rope_kernel(
    const float* __restrict__ qkv, const float* __restrict__ cs,
    const float* __restrict__ sn, float* __restrict__ Qr, float* __restrict__ Kr,
    float* __restrict__ Vr, int B, int T, int C, int H)
{
  const int HD = 64;
  const size_t idx = (size_t)blockIdx.x * 256 + threadIdx.x;
  if (idx >= (size_t)B * T * C) return;
  const int d = (int)(idx & 63);
  size_t q1 = idx >> 6;
  const int h = (int)(q1 % H); q1 /= H;
  const int t = (int)(q1 % T);
  const int b = (int)(q1 / T);
  const float* rowp = qkv + ((size_t)b * T + t) * (3 * C) + h * HD;
  const float qv = rowp[d], kv = rowp[C + d], vv = rowp[2 * C + d];
  const int d2 = (d < 32) ? d + 32 : d - 32;
  const float sgn = (d < 32) ? -1.f : 1.f;
  const float qrh = rowp[d2] * sgn, krh = rowp[C + d2] * sgn;
  const float cvv = cs[(size_t)t * HD + d], svv = sn[(size_t)t * HD + d];
  const size_t o = (((size_t)b * H + h) * T + t) * HD + d;
  Qr[o] = qv * cvv + qrh * svv;
  Kr[o] = kv * cvv + krh * svv;
  Vr[o] = vv;
}

// v_dyn = cg * (cv - v_ret)
__global__ __launch_bounds__(256) void vdyn_kernel(
    const float* __restrict__ cg, const float* __restrict__ cv,
    const float* __restrict__ vret, float* __restrict__ vd, long long n)
{
  const size_t i = (size_t)blockIdx.x * 256 + threadIdx.x;
  if (i < (size_t)n) vd[i] = cg[i] * (cv[i] - vret[i]);
}

// ---------------------------------------------------------------------------
// Causal flash attention, HD=64. One wave per 16 query rows per (b,h).
// Q,K,V: [B*H, T, 64] f32.  Y: [B,T,C] (head h writes cols h*64..h*64+63).
// grid = (T/16, B*H), block = 32.
// ---------------------------------------------------------------------------
__global__ __launch_bounds__(32) void attn_kernel(
    const float* __restrict__ Q, const float* __restrict__ Km,
    const float* __restrict__ V, float* __restrict__ Y, int T, int H, int C)
{
  const int HD = 64;
  const int lane = threadIdx.x;
  const int r = lane & 15;
  const int half = lane >> 4;
  const int bh = blockIdx.y;
  const int b = bh / H;
  const int h = bh % H;
  const int row0 = blockIdx.x << 4;

  const float* Qb = Q + (size_t)bh * T * HD;
  const float* Kb = Km + (size_t)bh * T * HD;
  const float* Vb = V + (size_t)bh * T * HD;

  v16h aq0 = loadA16(Qb, HD, row0 + r, 0, half);
  v16h aq1 = loadA16(Qb, HD, row0 + r, 32, half);

  float m_i[8], l_i[8];
#pragma unroll
  for (int i = 0; i < 8; ++i) { m_i[i] = -1e30f; l_i[i] = 0.f; }
  v8f o0 = {}, o1 = {}, o2 = {}, o3 = {};

  __shared__ float pls[256];  // 16x16 probability tile for C->A transpose

  for (int s0 = 0; s0 <= row0; s0 += 16) {
    v16h bk0 = loadB_nk(Kb, HD, s0 + r, 0, half);
    v16h bk1 = loadB_nk(Kb, HD, s0 + r, 32, half);
    v8f s = {};
    s = __builtin_amdgcn_wmma_f32_16x16x32_f16(false, aq0, false, bk0, (short)0, s, false, false);
    s = __builtin_amdgcn_wmma_f32_16x16x32_f16(false, aq1, false, bk1, (short)0, s, false, false);

    float sv[8];
#pragma unroll
    for (int i = 0; i < 8; ++i) {
      float v = s[i] * 0.125f;                  // 1/sqrt(64)
      const int rr = row0 + 8 * half + i;
      const int cc = s0 + r;
      if (cc > rr) v = -1e30f;                  // causal (diagonal tile only)
      sv[i] = v;
    }

    // online softmax; per-row reductions across each 16-lane half
#pragma unroll
    for (int i = 0; i < 8; ++i) {
      float mx = sv[i];
#pragma unroll
      for (int m = 8; m >= 1; m >>= 1) mx = fmaxf(mx, __shfl_xor(mx, m, 16));
      const float nm = fmaxf(mx, m_i[i]);
      const float corr = __expf(m_i[i] - nm);
      float p = __expf(sv[i] - nm);
      float ps = p;
#pragma unroll
      for (int m = 8; m >= 1; m >>= 1) ps += __shfl_xor(ps, m, 16);
      l_i[i] = l_i[i] * corr + ps;
      m_i[i] = nm;
      o0[i] *= corr; o1[i] *= corr; o2[i] *= corr; o3[i] *= corr;
      sv[i] = p;
    }

    __syncthreads();
#pragma unroll
    for (int i = 0; i < 8; ++i) pls[(8 * half + i) * 16 + r] = sv[i];
    __syncthreads();

    v16h ap;
#pragma unroll
    for (int j = 0; j < 16; ++j) {
      const int kj = 16 * half + j;
      ap[j] = (kj < 16) ? (_Float16)pls[r * 16 + kj] : (_Float16)0.f;
    }

#pragma unroll
    for (int g = 0; g < 4; ++g) {
      v16h bv;
      if (half == 0) {
#pragma unroll
        for (int j = 0; j < 16; ++j)
          bv[j] = (_Float16)Vb[(size_t)(s0 + j) * HD + g * 16 + r];
      } else {
#pragma unroll
        for (int j = 0; j < 16; ++j) bv[j] = (_Float16)0.f;
      }
      if (g == 0) o0 = __builtin_amdgcn_wmma_f32_16x16x32_f16(false, ap, false, bv, (short)0, o0, false, false);
      if (g == 1) o1 = __builtin_amdgcn_wmma_f32_16x16x32_f16(false, ap, false, bv, (short)0, o1, false, false);
      if (g == 2) o2 = __builtin_amdgcn_wmma_f32_16x16x32_f16(false, ap, false, bv, (short)0, o2, false, false);
      if (g == 3) o3 = __builtin_amdgcn_wmma_f32_16x16x32_f16(false, ap, false, bv, (short)0, o3, false, false);
    }
  }

  float* Yb = Y + (size_t)b * T * C + (size_t)h * HD;
#pragma unroll
  for (int i = 0; i < 8; ++i) {
    const float inv = 1.0f / l_i[i];
    float* yr = Yb + (size_t)(row0 + 8 * half + i) * C;
    yr[r]      = o0[i] * inv;
    yr[16 + r] = o1[i] * inv;
    yr[32 + r] = o2[i] * inv;
    yr[48 + r] = o3[i] * inv;
  }
}

// ---------------------------------------------------------------------------
// Host-side orchestration
// ---------------------------------------------------------------------------
extern "C" void kernel_launch(void* const* d_in, const int* in_sizes, int n_in,
                              void* d_out, int out_size, void* d_ws, size_t ws_size,
                              hipStream_t stream) {
  (void)in_sizes; (void)n_in; (void)out_size; (void)ws_size;
  constexpr int B = 2, T = 2048, C = 1024, H = 16, FH = 1536;
  constexpr int ROWS = B * T;                    // 4096
  const float scaleC = 0.03125f;                 // 1/sqrt(1024)
  const size_t N1 = (size_t)B * T * C;           // 4M

  const float* x       = (const float*)d_in[0];
  const float* cs      = (const float*)d_in[1];
  const float* sn      = (const float*)d_in[2];
  const float* qkv_w   = (const float*)d_in[3];
  const float* proj_w  = (const float*)d_in[4];
  const float* proj_b  = (const float*)d_in[5];
  const float* norm_g  = (const float*)d_in[6];
  const float* norm_b  = (const float*)d_in[7];
  const float* anorm_g = (const float*)d_in[8];
  const float* anorm_b = (const float*)d_in[9];
  const float* ffn1_w  = (const float*)d_in[10];
  const float* ffn1_b  = (const float*)d_in[11];
  const float* ffn2_w  = (const float*)d_in[12];
  const float* ffn2_b  = (const float*)d_in[13];
  const float* mem     = (const float*)d_in[14];
  // d_in[15..16] (memg_w/b) feed only the dead _next_mem path -> skipped
  const float* mnorm_g = (const float*)d_in[17];
  const float* mnorm_b = (const float*)d_in[18];
  const float* wq_w = (const float*)d_in[19]; const float* wq_b = (const float*)d_in[20];
  const float* wk_w = (const float*)d_in[21]; const float* wk_b = (const float*)d_in[22];
  const float* wv_w = (const float*)d_in[23]; const float* wv_b = (const float*)d_in[24];
  const float* wg_w = (const float*)d_in[25]; const float* wg_b = (const float*)d_in[26];
  const float* wo_w = (const float*)d_in[27]; const float* wo_b = (const float*)d_in[28];
  const float* op_w = (const float*)d_in[29]; const float* op_b = (const float*)d_in[30];

  // workspace layout (floats)
  float* ws  = (float*)d_ws;
  float* A   = ws;                                 // 4M  (LN out / attn Y / mc)
  float* F   = A + N1;                             // 6M  (ffn hidden)
  float* QKV = F + (size_t)B * T * FH;             // 12M (qkv; later cq/ck/cv; later u)
  float* Qr  = QKV + 3 * N1;                       // 4M
  float* Kr  = Qr + N1;                            // 4M
  float* Vr  = Kr + N1;                            // 4M
  float* XF  = Vr + N1;                            // 4M (x_ffn)
  float* S2  = XF + N1;                            // 4M (x_ffn + attn_ctx)
  float* MO  = S2 + N1;                            // 4M (mo)
  float* cq = QKV, *ck = QKV + N1, *cv = QKV + 2 * N1;
  float* cg = Qr, *vret = Kr, *vd = Vr;
  float* U  = QKV;                                 // 16M (reuses QKV + Qr region)

  float* outp = (float*)d_out;
  float* scc  = outp + N1;                         // [B,T,T]

  const long long z = 0;
  dim3 b128(128), b256(256);

  // 1) h1 = LN(x)
  ln_kernel<<<ROWS, b256, 0, stream>>>(x, norm_g, norm_b, A, C);
  // 2) F = silu(h1 @ ffn1^T + b1)
  gemm_wmma<0, 1, true, false, false><<<dim3(FH / 128, ROWS / 64, 1), b128, 0, stream>>>(
      A, z, C, ffn1_w, z, C, ffn1_b, nullptr, z, 0, F, z, FH, C, 1.f);
  // 3) x_ffn = F @ ffn2^T + b2 + x
  gemm_wmma<0, 0, true, true, false><<<dim3(C / 128, ROWS / 64, 1), b128, 0, stream>>>(
      F, z, FH, ffn2_w, z, FH, ffn2_b, x, z, C, XF, z, C, FH, 1.f);
  // 4) h = LN(x_ffn)
  ln_kernel<<<ROWS, b256, 0, stream>>>(XF, anorm_g, anorm_b, A, C);
  // 5) qkv = h @ qkv_w^T
  gemm_wmma<0, 0, false, false, false><<<dim3(3 * C / 128, ROWS / 64, 1), b128, 0, stream>>>(
      A, z, C, qkv_w, z, C, nullptr, nullptr, z, 0, QKV, z, 3 * C, C, 1.f);
  // 6) RoPE + head split
  rope_kernel<<<(unsigned)((N1 + 255) / 256), b256, 0, stream>>>(
      QKV, cs, sn, Qr, Kr, Vr, B, T, C, H);
  // 7) causal attention -> Y (stored in A)
  attn_kernel<<<dim3(T / 16, B * H), dim3(32), 0, stream>>>(Qr, Kr, Vr, A, T, H, C);
  // 8) s2 = Y @ proj^T + proj_b + x_ffn
  gemm_wmma<0, 0, true, true, false><<<dim3(C / 128, ROWS / 64, 1), b128, 0, stream>>>(
      A, z, C, proj_w, z, C, proj_b, XF, z, C, S2, z, C, C, 1.f);
  // 9) mc = LN(s2)
  ln_kernel<<<ROWS, b256, 0, stream>>>(S2, mnorm_g, mnorm_b, A, C);
  // 10-13) memory-branch projections
  gemm_wmma<0, 1, true, false, false><<<dim3(C / 128, ROWS / 64, 1), b128, 0, stream>>>(
      A, z, C, wq_w, z, C, wq_b, nullptr, z, 0, cq, z, C, C, 1.f);            // cq=silu
  gemm_wmma<0, 0, true, false, false><<<dim3(C / 128, ROWS / 64, 1), b128, 0, stream>>>(
      A, z, C, wk_w, z, C, wk_b, nullptr, z, 0, ck, z, C, C, 1.f);            // ck
  rownorm_kernel<<<ROWS, b256, 0, stream>>>(ck, C);
  gemm_wmma<0, 0, true, false, false><<<dim3(C / 128, ROWS / 64, 1), b128, 0, stream>>>(
      A, z, C, wv_w, z, C, wv_b, nullptr, z, 0, cv, z, C, C, 1.f);            // cv
  gemm_wmma<0, 2, true, false, false><<<dim3(C / 128, ROWS / 64, 1), b128, 0, stream>>>(
      A, z, C, wg_w, z, C, wg_b, nullptr, z, 0, cg, z, C, C, 1.f);            // cg=sigmoid
  // 14) v_ret = scale * ck @ mem   (mem is [K,N], batch-broadcast)
  gemm_wmma<1, 0, false, false, false><<<dim3(C / 128, ROWS / 64, 1), b128, 0, stream>>>(
      ck, z, C, mem, z, C, nullptr, nullptr, z, 0, vret, z, C, C, scaleC);
  // 15) v_dyn = cg * (cv - v_ret)
  vdyn_kernel<<<(unsigned)((N1 + 255) / 256), b256, 0, stream>>>(cg, cv, vret, vd, (long long)N1);
  // 16) sc_c = tril(cq @ ck^T)  -> d_out second region (per batch)
  gemm_wmma<0, 0, false, false, true><<<dim3(T / 128, T / 64, B), b128, 0, stream>>>(
      cq, (long long)T * C, C, ck, (long long)T * C, C, nullptr,
      nullptr, z, 0, scc, (long long)T * T, T, C, 1.f);
  // 17) mo = scale * cq @ mem
  gemm_wmma<1, 0, false, false, false><<<dim3(C / 128, ROWS / 64, 1), b128, 0, stream>>>(
      cq, z, C, mem, z, C, nullptr, nullptr, z, 0, MO, z, C, C, scaleC);
  // 18) mo += scale * sc_c @ v_dyn  (per batch; v_dyn is [K,N])
  gemm_wmma<1, 0, false, true, false><<<dim3(C / 128, T / 64, B), b128, 0, stream>>>(
      scc, (long long)T * T, T, vd, (long long)T * C, C, nullptr,
      MO, (long long)T * C, C, MO, (long long)T * C, C, T, scaleC);
  // 19) u = silu(mo @ wo^T + wo_b)
  gemm_wmma<0, 1, true, false, false><<<dim3(4 * C / 128, ROWS / 64, 1), b128, 0, stream>>>(
      MO, z, C, wo_w, z, C, wo_b, nullptr, z, 0, U, z, 4 * C, C, 1.f);
  // 20) out = u @ op^T + op_b + s2
  gemm_wmma<0, 0, true, true, false><<<dim3(C / 128, ROWS / 64, 1), b128, 0, stream>>>(
      U, z, 4 * C, op_w, z, 4 * C, op_b, S2, z, C, outp, z, C, 4 * C, 1.f);
}
